// FastRNNlayer_83932250898452
// MI455X (gfx1250) — compile-verified, matrix operand order checked by reading, not run on hardware
//
#include <hip/hip_runtime.h>

// Shapes
#define SEQ 512
#define BSZ 32
#define DIM 256
#define NH  8
#define DH  32
#define QKV_E (NH * (5 * DH + 2))   // 1296
#define SB  (SEQ * BSZ)             // 16384
#define LN_EPS 1e-5f

typedef float    v8f  __attribute__((ext_vector_type(8)));
typedef _Float16 v8h  __attribute__((ext_vector_type(8)));
typedef _Float16 v16h __attribute__((ext_vector_type(16)));

// ---------------- wave32 helpers ----------------
__device__ __forceinline__ float wsum32(float v) {
#pragma unroll
    for (int m = 16; m > 0; m >>= 1) v += __shfl_xor(v, m, 32);
    return v;
}
__device__ __forceinline__ float wmax32(float v) {
#pragma unroll
    for (int m = 16; m > 0; m >>= 1) v = fmaxf(v, __shfl_xor(v, m, 32));
    return v;
}
// Broadcast lane j's value to all lanes via v_readlane -> SGPR operand of FMA
__device__ __forceinline__ float bcast(float v, int j) {
    return __int_as_float(__builtin_amdgcn_readlane(__float_as_int(v), j));
}
__device__ __forceinline__ float sigmoidf_(float x) { return 1.0f / (1.0f + __expf(-x)); }
__device__ __forceinline__ float elu_p1(float x) { return x > 0.0f ? x + 1.0f : __expf(x); }

// 32-wide matvec: lane i owns row i; vec element j broadcast from lane j.
// 4 accumulator trees -> dependent-FMA chain depth 8 instead of 32.
__device__ __forceinline__ float matvec32(const float* __restrict__ row, float vec) {
    float a0 = 0.f, a1 = 0.f, a2 = 0.f, a3 = 0.f;
#pragma unroll
    for (int j = 0; j < DH; j += 4) {
        a0 = __builtin_fmaf(row[j + 0], bcast(vec, j + 0), a0);
        a1 = __builtin_fmaf(row[j + 1], bcast(vec, j + 1), a1);
        a2 = __builtin_fmaf(row[j + 2], bcast(vec, j + 2), a2);
        a3 = __builtin_fmaf(row[j + 3], bcast(vec, j + 3), a3);
    }
    return (a0 + a1) + (a2 + a3);
}
// Rank-1 update row[j] += d * vec[j]; fully independent FMAs.
__device__ __forceinline__ void rank1(float* __restrict__ row, float d, float vec) {
#pragma unroll
    for (int j = 0; j < DH; ++j) row[j] = __builtin_fmaf(d, bcast(vec, j), row[j]);
}

// ---------------- Kernel 1: LayerNorm -> f16 ----------------
// One wave per row of 256; 8 waves per block.
__global__ void ln_kernel(const float* __restrict__ x, const float* __restrict__ g,
                          const float* __restrict__ b, _Float16* __restrict__ o16) {
    int row  = blockIdx.x * 8 + (threadIdx.x >> 5);
    int lane = threadIdx.x & 31;
    const float* xr = x + (size_t)row * DIM;
    float vals[8];
    float s = 0.f, ss = 0.f;
#pragma unroll
    for (int i = 0; i < 8; ++i) {
        float v = xr[lane + i * 32];
        vals[i] = v; s += v; ss += v * v;
    }
    s = wsum32(s); ss = wsum32(ss);
    float mu = s * (1.0f / DIM);
    float var = ss * (1.0f / DIM) - mu * mu;
    float rstd = rsqrtf(var + LN_EPS);
    _Float16* orow = o16 + (size_t)row * DIM;
#pragma unroll
    for (int i = 0; i < 8; ++i) {
        int d = lane + i * 32;
        orow[d] = (_Float16)((vals[i] - mu) * rstd * g[d] + b[d]);
    }
}

// ---------------- Kernel 2: f32 -> f16 convert ----------------
__global__ void cvt_kernel(const float* __restrict__ src, _Float16* __restrict__ dst, int n) {
    int i = blockIdx.x * blockDim.x + threadIdx.x;
    if (i < n) dst[i] = (_Float16)src[i];
}

// ---------------- WMMA fragment load ----------------
// 16-bit A/B 16x32 fragment, ISA layout: lanes 0-15 hold K=k0+0..7 and k0+16..23
// of row (lane&15); lanes 16-31 hold K=k0+8..15 and k0+24..31.
__device__ __forceinline__ v16h load_frag(const _Float16* __restrict__ row, int k0, int half) {
    v8h lo = *(const v8h*)(row + k0 + half * 8);
    v8h hi = *(const v8h*)(row + k0 + 16 + half * 8);
    v16h r;
#pragma unroll
    for (int i = 0; i < 8; ++i) { r[i] = lo[i]; r[i + 8] = hi[i]; }
    return r;
}

// ---------------- Kernel 3: qkv GEMM ----------------
// C[m,e] = sum_d A[m,d]*W[e,d];  M=16384, N=1296, K=256. One 16x16 tile per wave.
__global__ void gemm_qkv(const _Float16* __restrict__ A, const _Float16* __restrict__ W,
                         float* __restrict__ C) {
    int wid  = (blockIdx.x * blockDim.x + threadIdx.x) >> 5;
    int lane = threadIdx.x & 31;
    const int NT = QKV_E / 16;           // 81
    int tm = wid / NT, tn = wid % NT;
    if (tm >= SB / 16) return;
    int half = lane >> 4, r = lane & 15;
    const _Float16* arow = A + (size_t)(tm * 16 + r) * DIM;
    const _Float16* brow = W + (size_t)(tn * 16 + r) * DIM;
    v8f acc = {};
#pragma unroll
    for (int k0 = 0; k0 < DIM; k0 += 32) {
        v16h a = load_frag(arow, k0, half);
        v16h b = load_frag(brow, k0, half);
        acc = __builtin_amdgcn_wmma_f32_16x16x32_f16(false, a, false, b, (short)0, acc, false, false);
    }
#pragma unroll
    for (int v = 0; v < 8; ++v) {
        int m = tm * 16 + v + 8 * half;          // C layout: VGPR v, lanes0-15 M=v, lanes16-31 M=v+8
        int n = tn * 16 + r;
        C[(size_t)m * QKV_E + n] = acc[v];
    }
}

// ---------------- Kernel 4: fused nonlinearities + dual fast-weight scan ----------------
// One wave per (b,h). Lane i owns row i of W and R (32 VGPRs each). Serial over S.
// Next step's inputs are software-prefetched to overlap memory latency with the
// ~400-op serial compute body.
__global__ void scan_kernel(const float* __restrict__ qkv, _Float16* __restrict__ hs16) {
    int bh   = blockIdx.x;            // 0..255
    int b    = bh >> 3;
    int hh   = bh & 7;
    int lane = threadIdx.x & 31;

    float Wrow[DH], Rrow[DH];
#pragma unroll
    for (int j = 0; j < DH; ++j) { Wrow[j] = 0.f; Rrow[j] = 0.f; }
    float h = 0.f;

    const size_t sstride = (size_t)BSZ * QKV_E;      // per-seq-step stride
    const float* base = qkv + (size_t)b * QKV_E + hh * (5 * DH + 2);

    // preload step 0
    float q  = base[lane];
    float k  = base[DH + lane];
    float v  = base[2 * DH + lane];
    float rk = base[3 * DH + lane];
    float rv = base[4 * DH + lane];
    float braw  = base[5 * DH];
    float rbraw = base[5 * DH + 1];

    for (int s = 0; s < SEQ; ++s) {
        // ---- prefetch step s+1 (issues loads, consumed at loop bottom) ----
        const float* nb = base + sstride;
        float qn = 0.f, kn = 0.f, vn = 0.f, rkn = 0.f, rvn = 0.f, bn = 0.f, rbn = 0.f;
        if (s + 1 < SEQ) {
            qn  = nb[lane];
            kn  = nb[DH + lane];
            vn  = nb[2 * DH + lane];
            rkn = nb[3 * DH + lane];
            rvn = nb[4 * DH + lane];
            bn  = nb[5 * DH];
            rbn = nb[5 * DH + 1];
        }

        float beta  = sigmoidf_(braw);
        float rbeta = sigmoidf_(rbraw);

        // q,k: sum_norm(elu+1); rk: softmax over DH
        float qv = elu_p1(q);  qv /= wsum32(qv);
        float kv = elu_p1(k);  kv /= wsum32(kv);
        float rkv = __expf(rk - wmax32(rk));
        rkv /= wsum32(rkv);

        // ---- feed-forward fast weights (delta rule) ----
        float vo = matvec32(Wrow, kv);
        float d  = beta * (v - vo);
        rank1(Wrow, d, kv);
        float z = matvec32(Wrow, qv);

        // ---- recurrent fast weights: query = softmax(prev h) ----
        float qe = __expf(h - wmax32(h));
        float qr = qe / wsum32(qe);
        float vor = matvec32(Rrow, rkv);
        float dr  = rbeta * (rv - vor);
        rank1(Rrow, dr, rkv);
        float hr = matvec32(Rrow, qr);
        h = z + hr;

        hs16[((size_t)(s * BSZ + b)) * (NH * DH) + hh * DH + lane] = (_Float16)h;

        // rotate prefetched inputs
        q = qn; k = kn; v = vn; rk = rkn; rv = rvn; braw = bn; rbraw = rbn;
        base = nb;
    }
}

// ---------------- Kernel 5: output GEMM + residual ----------------
// out[m,d] = x[m,d] + sum_e hs[m,e]*outW[d,e];  M=16384, N=256, K=256.
__global__ void gemm_out(const _Float16* __restrict__ A, const _Float16* __restrict__ W,
                         const float* __restrict__ x, float* __restrict__ out) {
    int wid  = (blockIdx.x * blockDim.x + threadIdx.x) >> 5;
    int lane = threadIdx.x & 31;
    const int NT = DIM / 16;             // 16
    int tm = wid / NT, tn = wid % NT;
    if (tm >= SB / 16) return;
    int half = lane >> 4, r = lane & 15;
    const _Float16* arow = A + (size_t)(tm * 16 + r) * DIM;
    const _Float16* brow = W + (size_t)(tn * 16 + r) * DIM;
    v8f acc = {};
#pragma unroll
    for (int k0 = 0; k0 < DIM; k0 += 32) {
        v16h a = load_frag(arow, k0, half);
        v16h b = load_frag(brow, k0, half);
        acc = __builtin_amdgcn_wmma_f32_16x16x32_f16(false, a, false, b, (short)0, acc, false, false);
    }
#pragma unroll
    for (int v = 0; v < 8; ++v) {
        int m = tm * 16 + v + 8 * half;
        int n = tn * 16 + r;
        size_t idx = (size_t)m * DIM + n;
        out[idx] = x[idx] + acc[v];
    }
}

// ---------------- launch ----------------
extern "C" void kernel_launch(void* const* d_in, const int* in_sizes, int n_in,
                              void* d_out, int out_size, void* d_ws, size_t ws_size,
                              hipStream_t stream) {
    const float* x      = (const float*)d_in[0];  // [512,32,256]
    const float* slow_W = (const float*)d_in[1];  // [1296,256]
    const float* out_W  = (const float*)d_in[2];  // [256,256]
    const float* ln_g   = (const float*)d_in[3];  // [256]
    const float* ln_b   = (const float*)d_in[4];  // [256]
    float* out = (float*)d_out;

    char* ws = (char*)d_ws;
    // workspace layout (all offsets 256B-aligned)
    const size_t OFF_O16  = 0;                                   // 16384*256 f16 = 8 MB
    const size_t OFF_SW16 = OFF_O16  + (size_t)SB * DIM * 2;     // 1296*256 f16
    const size_t OFF_OW16 = OFF_SW16 + (size_t)QKV_E * DIM * 2;  // 256*256 f16
    const size_t OFF_QKV  = OFF_OW16 + (size_t)DIM * DIM * 2;    // 16384*1296 f32 = 85 MB
    const size_t OFF_HS16 = OFF_QKV  + (size_t)SB * QKV_E * 4;   // 16384*256 f16 = 8 MB

    _Float16* o16  = (_Float16*)(ws + OFF_O16);
    _Float16* sw16 = (_Float16*)(ws + OFF_SW16);
    _Float16* ow16 = (_Float16*)(ws + OFF_OW16);
    float*    qkv  = (float*)   (ws + OFF_QKV);
    _Float16* hs16 = (_Float16*)(ws + OFF_HS16);

    // 1. LayerNorm (f32 -> f16 activations)
    ln_kernel<<<SB / 8, 256, 0, stream>>>(x, ln_g, ln_b, o16);
    // 2. weight converts
    cvt_kernel<<<(QKV_E * DIM + 255) / 256, 256, 0, stream>>>(slow_W, sw16, QKV_E * DIM);
    cvt_kernel<<<(DIM * DIM + 255) / 256, 256, 0, stream>>>(out_W, ow16, DIM * DIM);
    // 3. qkv projection: 1024 x 81 tiles of 16x16, one per wave, 8 waves/block
    gemm_qkv<<<(SB / 16) * (QKV_E / 16) / 8, 256, 0, stream>>>(o16, sw16, qkv);
    // 4. fused nonlinearity + serial dual fast-weight scan: one wave per (b,h)
    scan_kernel<<<BSZ * NH, 32, 0, stream>>>(qkv, hs16);
    // 5. output projection + residual: 1024 x 16 tiles
    gemm_out<<<(SB / 16) * (DIM / 16) / 8, 256, 0, stream>>>(hs16, ow16, x, out);
}
